// KarankEtAl2_88493506167122
// MI455X (gfx1250) — compile-verified, matrix operand order checked by reading, not run on hardware
//
#include <hip/hip_runtime.h>
#include <hip/hip_bf16.h>

typedef __attribute__((ext_vector_type(16))) _Float16 v16h;
typedef __attribute__((ext_vector_type(8)))  _Float16 v8h;
typedef __attribute__((ext_vector_type(8)))  float    v8f;

#define B_BATCH 16384
#define NCOMP   12
#define NCLS    16
#define C2      108           // 9*NCOMP
#define FEAT    1296          // C2*NCOMP
#define FC1OUT  72            // 6*NCOMP
#define KK2     11            // conv2 K steps (324 -> 352 = 11*32)
#define NT2     7             // conv2 N tiles (108 -> 112 = 7*16)
#define KPAD3   1312          // fc1 K padded (1296 -> 1312 = 41*32)
#define KK3     41            // fc1 K steps
#define NT3     5             // fc1 N tiles (72 -> 80 = 5*16)

// Fragment element -> (k,n) mapping for V_WMMA_F32_16X16X32_F16 B operand:
//   lane: n = nt*16 + (lane&15), half = lane>>4
//   elem e: v=e>>1, p=e&1, grp=v>>2, vv=v&3
//   k = kb + grp*16 + half*8 + vv*2 + p
__device__ __forceinline__ int frag_k(int kb, int half, int e) {
    int v = e >> 1, p = e & 1;
    int grp = v >> 2, vv = v & 3;
    return kb + grp * 16 + half * 8 + vv * 2 + p;
}

// ---------------------------------------------------------------------------
// Kernel 0: pre-swizzle weights into per-lane WMMA B fragments (f16, padded).
//   w2p : [nt=7][kk=11][lane=32][e=16]  from conv2_w (B[k][n] = w2[n*324+k])
//   fc1p: [nt=5][kk=41][lane=32][e=16]  from fc1_w   (B[k][n] = fc1w[n*1296+k])
// ---------------------------------------------------------------------------
__global__ __launch_bounds__(256) void prep_weights(
    const float* __restrict__ w2, const float* __restrict__ fc1w,
    _Float16* __restrict__ w2p, _Float16* __restrict__ fc1p)
{
    const int t = blockIdx.x * 256 + threadIdx.x;
    const int n2 = NT2 * KK2 * 32 * 16;     // 39424
    const int n3 = NT3 * KK3 * 32 * 16;     // 104960
    if (t < n2) {
        int e    = t & 15;
        int lane = (t >> 4) & 31;
        int kk   = (t >> 9) % KK2;
        int nt   = t / (512 * KK2);
        int n    = nt * 16 + (lane & 15);
        int k    = frag_k(kk * 32, lane >> 4, e);
        float v  = (k < 324 && n < C2) ? w2[n * 324 + k] : 0.f;
        w2p[t] = (_Float16)v;
    } else if (t < n2 + n3) {
        int t2   = t - n2;
        int e    = t2 & 15;
        int lane = (t2 >> 4) & 31;
        int kk   = (t2 >> 9) % KK3;
        int nt   = t2 / (512 * KK3);
        int n    = nt * 16 + (lane & 15);
        int k    = frag_k(kk * 32, lane >> 4, e);
        float v  = (k < FEAT && n < FC1OUT) ? fc1w[n * FEAT + k] : 0.f;
        fc1p[t2] = (_Float16)v;
    }
}

// ---------------------------------------------------------------------------
// Kernel 1: per-sample PCA (one 256-thread WG per sample, LDS resident).
// ---------------------------------------------------------------------------
__global__ __launch_bounds__(256) void pca_kernel(
    const float* __restrict__ x, float* __restrict__ z)
{
    __shared__ float y[5000];
    __shared__ float mu[200];
    __shared__ float G[625];
    __shared__ float U[625];
    __shared__ float M[625];
    __shared__ float shc[2];
    __shared__ int   topIdx[12];
    __shared__ float invS[12];

    const int b   = blockIdx.x;
    const int tid = threadIdx.x;
    const float* xb = x + (size_t)b * 5000;

    for (int t = tid; t < 5000; t += 256) {
        if (t + 1280 < 5000) __builtin_prefetch(&xb[t + 1280], 0, 1);
        y[t] = xb[t];
    }
    __syncthreads();

    for (int t = tid; t < 200; t += 256) {
        float m = 0.f;
        for (int r = 0; r < 25; ++r) m += y[r * 200 + t];
        mu[t] = m * (1.f / 25.f);
    }
    __syncthreads();

    for (int e = tid; e < 625; e += 256) {
        int i = e / 25, j = e % 25;
        float g = 0.f;
        for (int s = 0; s < 200; ++s)
            g += (y[i * 200 + s] - mu[s]) * (y[j * 200 + s] - mu[s]);
        G[e] = g;
        U[e] = (i == j) ? 1.f : 0.f;
    }
    __syncthreads();

    // Cyclic Jacobi eigensolve of symmetric 25x25 G, accumulate U.
    for (int sweep = 0; sweep < 7; ++sweep) {
        for (int p = 0; p < 24; ++p) {
            for (int q = p + 1; q < 25; ++q) {
                if (tid == 0) {
                    float apq = G[p * 25 + q];
                    float app = G[p * 25 + p], aqq = G[q * 25 + q];
                    float c = 1.f, sn = 0.f;
                    if (fabsf(apq) > 1e-12f) {
                        float th = 0.5f * (aqq - app) / apq;
                        float tt = ((th >= 0.f) ? 1.f : -1.f) /
                                   (fabsf(th) + sqrtf(th * th + 1.f));
                        c = 1.f / sqrtf(tt * tt + 1.f);
                        sn = tt * c;
                    }
                    shc[0] = c; shc[1] = sn;
                }
                __syncthreads();
                const float c = shc[0], sn = shc[1];
                if (tid < 25) {
                    const int k = tid;
                    float gkp = G[k * 25 + p], gkq = G[k * 25 + q];
                    G[k * 25 + p] = c * gkp - sn * gkq;
                    G[k * 25 + q] = sn * gkp + c * gkq;
                    float ukp = U[k * 25 + p], ukq = U[k * 25 + q];
                    U[k * 25 + p] = c * ukp - sn * ukq;
                    U[k * 25 + q] = sn * ukp + c * ukq;
                }
                __syncthreads();
                if (tid < 25) {
                    const int k = tid;
                    float gpk = G[p * 25 + k], gqk = G[q * 25 + k];
                    G[p * 25 + k] = c * gpk - sn * gqk;
                    G[q * 25 + k] = sn * gpk + c * gqk;
                }
                __syncthreads();
            }
        }
    }

    if (tid == 0) {
        float ev[25]; int id[25];
        for (int k = 0; k < 25; ++k) { ev[k] = G[k * 25 + k]; id[k] = k; }
        for (int j = 0; j < 12; ++j) {
            int best = j;
            for (int k = j + 1; k < 25; ++k) if (ev[k] > ev[best]) best = k;
            float tv = ev[j]; ev[j] = ev[best]; ev[best] = tv;
            int ti = id[j]; id[j] = id[best]; id[best] = ti;
            topIdx[j] = id[j];
            invS[j] = 1.f / sqrtf(fmaxf(ev[j], 1e-20f));
        }
    }
    __syncthreads();

    // M = y * yc^T  (projects UNcentered y, as in the reference)
    for (int e = tid; e < 625; e += 256) {
        int i = e / 25, j = e % 25;
        float m = 0.f;
        for (int s = 0; s < 200; ++s)
            m += y[i * 200 + s] * (y[j * 200 + s] - mu[s]);
        M[e] = m;
    }
    __syncthreads();

    float* zb = z + (size_t)b * 300;
    for (int e = tid; e < 300; e += 256) {
        int r = e / 12, j = e % 12;
        int col = topIdx[j];
        float pv = 0.f;
        for (int k = 0; k < 25; ++k) pv += M[r * 25 + k] * U[k * 25 + col];
        zb[e] = pv * invS[j];
    }
}

// ---------------------------------------------------------------------------
// Kernel 2: fused conv1 (VALU) + conv2 (WMMA).  One WG per sample.
//   A (LDS, f16): A[d][i*9+q], 16x352 zero-padded.
//   Waves 0..6 own one 16-col N-tile; 11 fully-unrolled WMMA K-steps.
//   Result staged in LDS, then one coalesced uint4 row store per sample.
// ---------------------------------------------------------------------------
__global__ __launch_bounds__(256) void conv_kernel(
    const float* __restrict__ z,
    const float* __restrict__ w1, const float* __restrict__ b1,
    const _Float16* __restrict__ w2p, const float* __restrict__ b2,
    _Float16* __restrict__ a2)
{
    __shared__ float zin[300];
    __shared__ float w1s[324];
    __shared__ float b1s[36];
    __shared__ __align__(16) _Float16 Amat[16 * KK2 * 32];   // 16 x 352
    __shared__ __align__(16) _Float16 a2s[KPAD3];            // 1312

    const int b   = blockIdx.x;
    const int tid = threadIdx.x;

    for (int t = tid; t < 300; t += 256) zin[t] = z[(size_t)b * 300 + t];
    for (int t = tid; t < 324; t += 256) w1s[t] = w1[t];
    if (tid < 36) b1s[tid] = b1[tid];
    for (int t = tid; t < 16 * 352 / 2; t += 256)
        ((unsigned int*)Amat)[t] = 0u;
    __syncthreads();

    for (int idx = tid; idx < 3888; idx += 256) {
        int i = idx / 108;
        int r = idx % 108;
        int d = r / 9;
        int q = r % 9;
        int ry = q / 3, rx = q % 3;
        float acc = b1s[i];
        #pragma unroll
        for (int ky = 0; ky < 3; ++ky)
            #pragma unroll
            for (int kx = 0; kx < 3; ++kx)
                acc += w1s[i * 9 + ky * 3 + kx] *
                       zin[d * 25 + (ry + ky) * 5 + (rx + kx)];
        Amat[d * 352 + i * 9 + q] = (_Float16)fmaxf(acc, 0.f);
    }
    __syncthreads();

    const int wave = tid >> 5;
    const int lane = tid & 31;
    if (wave < NT2) {                    // wave-uniform branch: EXEC all-1s
        const int nt   = wave;
        const int mrow = lane & 15;
        const int half = lane >> 4;
        const int ncol = nt * 16 + mrow;
        v8f acc = {};
        #pragma unroll
        for (int kk = 0; kk < KK2; ++kk) {
            const int kb = kk * 32;
            const _Float16* ap0 = &Amat[mrow * 352 + kb + half * 8];
            v8h lo = *(const v8h*)(ap0);
            v8h hi = *(const v8h*)(ap0 + 16);
            v16h a = __builtin_shufflevector(lo, hi, 0, 1, 2, 3, 4, 5, 6, 7,
                                             8, 9, 10, 11, 12, 13, 14, 15);
            v16h bm = *(const v16h*)(w2p + ((nt * KK2 + kk) * 32 + lane) * 16);
            acc = __builtin_amdgcn_wmma_f32_16x16x32_f16(
                false, a, false, bm, (short)0, acc, false, false);
        }
        if (ncol < C2) {
            const float bias = b2[ncol];
            #pragma unroll
            for (int v = 0; v < 8; ++v) {
                int mm = v + 8 * half;       // D: VGPR v, lanes16-31 -> M+8
                if (mm < NCOMP)
                    a2s[ncol * NCOMP + mm] =
                        (_Float16)fmaxf(acc[v] + bias, 0.f);
            }
        }
    }
    if (tid >= 224 && tid < 240) a2s[FEAT + (tid - 224)] = (_Float16)0.f;
    __syncthreads();

    // coalesced 16B row store: 1312 halves = 164 uint4
    const uint4* src = (const uint4*)a2s;
    uint4* dst = (uint4*)(a2 + (size_t)b * KPAD3);
    for (int t = tid; t < KPAD3 * 2 / 16; t += 256) dst[t] = src[t];
}

// ---------------------------------------------------------------------------
// Kernel 3: fused fc1 (WMMA) + fc2 (VALU).  One WG per 16 samples.
//   Waves 0..4 own one 16-col N-tile; 41 fully-unrolled WMMA K-steps.
// ---------------------------------------------------------------------------
__global__ __launch_bounds__(256) void fc_kernel(
    const _Float16* __restrict__ a2,
    const _Float16* __restrict__ fc1p,
    const float* __restrict__ fc1b,
    const float* __restrict__ fc2w,
    const float* __restrict__ fc2b,
    float* __restrict__ out)
{
    __shared__ float o1[16][80];

    const int tid  = threadIdx.x;
    const int wave = tid >> 5;
    const int lane = tid & 31;
    const int m0   = blockIdx.x * 16;

    if (wave < NT3) {                    // wave-uniform branch: EXEC all-1s
        const int nt   = wave;
        const int mrow = lane & 15;
        const int half = lane >> 4;
        const int ncol = nt * 16 + mrow;
        const _Float16* arow = a2 + (size_t)(m0 + mrow) * KPAD3;
        v8f acc = {};
        #pragma unroll
        for (int kk = 0; kk < KK3; ++kk) {
            const int kb = kk * 32;
            const _Float16* ap0 = arow + kb + half * 8;
            v8h lo = *(const v8h*)(ap0);
            v8h hi = *(const v8h*)(ap0 + 16);
            v16h a = __builtin_shufflevector(lo, hi, 0, 1, 2, 3, 4, 5, 6, 7,
                                             8, 9, 10, 11, 12, 13, 14, 15);
            v16h bm = *(const v16h*)(fc1p + ((nt * KK3 + kk) * 32 + lane) * 16);
            acc = __builtin_amdgcn_wmma_f32_16x16x32_f16(
                false, a, false, bm, (short)0, acc, false, false);
        }
        const float bias = (ncol < FC1OUT) ? fc1b[ncol] : 0.f;
        #pragma unroll
        for (int v = 0; v < 8; ++v) {
            int mm = v + 8 * half;
            o1[mm][ncol] = acc[v] + bias;
        }
    }
    __syncthreads();

    const int srow = tid >> 4;
    const int cls  = tid & 15;
    float o = fc2b[cls];
    #pragma unroll 8
    for (int j = 0; j < FC1OUT; ++j) o += fc2w[cls * FC1OUT + j] * o1[srow][j];
    out[(size_t)(m0 + srow) * NCLS + cls] = o;
}

// ---------------------------------------------------------------------------
extern "C" void kernel_launch(void* const* d_in, const int* in_sizes, int n_in,
                              void* d_out, int out_size, void* d_ws, size_t ws_size,
                              hipStream_t stream)
{
    const float* x    = (const float*)d_in[0];
    const float* w1   = (const float*)d_in[1];
    const float* b1   = (const float*)d_in[2];
    const float* w2   = (const float*)d_in[3];
    const float* b2   = (const float*)d_in[4];
    const float* fc1w = (const float*)d_in[5];
    const float* fc1b = (const float*)d_in[6];
    const float* fc2w = (const float*)d_in[7];
    const float* fc2b = (const float*)d_in[8];
    (void)in_sizes; (void)n_in; (void)out_size; (void)ws_size;

    char* ws = (char*)d_ws;
    float* z = (float*)ws;                                   // B*300 f32
    size_t off = (size_t)B_BATCH * 300 * sizeof(float);
    _Float16* a2 = (_Float16*)(ws + off);                    // B*1312 f16
    off += (size_t)B_BATCH * KPAD3 * sizeof(_Float16);
    _Float16* w2p = (_Float16*)(ws + off);                   // 39424 f16
    off += (size_t)NT2 * KK2 * 32 * 16 * sizeof(_Float16);
    _Float16* fc1p = (_Float16*)(ws + off);                  // 104960 f16

    const int prepN = NT2 * KK2 * 512 + NT3 * KK3 * 512;
    prep_weights<<<(prepN + 255) / 256, 256, 0, stream>>>(w2, fc1w, w2p, fc1p);
    pca_kernel<<<B_BATCH, 256, 0, stream>>>(x, z);
    conv_kernel<<<B_BATCH, 256, 0, stream>>>(z, w1, b1, w2p, b2, a2);
    fc_kernel<<<B_BATCH / 16, 256, 0, stream>>>(a2, fc1p, fc1b, fc2w, fc2b,
                                                (float*)d_out);
}
